// GNN_66949950210692
// MI455X (gfx1250) — matrix operand order repaired and checked by reading, not compiled, not run on hardware
//
#include <hip/hip_runtime.h>
#include <hip/hip_bf16.h>

#define N_NODES 40000
#define N_EDGES 640000
#define SD      128
#define N_GRAPHS 256
#define N_CLASS 41

typedef __attribute__((ext_vector_type(16))) __bf16 v16bf;
typedef __attribute__((ext_vector_type(8)))  float  v8f;

// ---------------------------------------------------------------- embedding
__global__ __launch_bounds__(256) void k_embed(const int* __restrict__ tok,
                                               const float* __restrict__ emb,
                                               float* __restrict__ X) {
    int t = blockIdx.x * 256 + threadIdx.x;      // N_NODES*32 threads
    int node = t >> 5, lane = t & 31;
    if (node < N_NODES) {
        float4 v = *(const float4*)(emb + (size_t)tok[node] * SD + lane * 4);
        *(float4*)(X + (size_t)node * SD + lane * 4) = v;
    }
}

// ------------------------------------------------- weight f32 -> bf16, transposed [N][K]
__global__ __launch_bounds__(256) void k_wconv(const float* __restrict__ W,
                                               __bf16* __restrict__ Wt) {
    int i = blockIdx.x * 256 + threadIdx.x;      // SD*SD threads
    int k = i >> 7, n = i & 127;
    if (i < SD * SD) Wt[n * SD + k] = (__bf16)W[k * SD + n];
}

// ---------------------------------------------------------------- copy x -> aggr
__global__ __launch_bounds__(256) void k_copy(const float* __restrict__ src,
                                              float* __restrict__ dst) {
    int i = blockIdx.x * 256 + threadIdx.x;      // N_NODES*SD/4 threads
    if (i < N_NODES * SD / 4)
        *(float4*)(dst + i * 4) = *(const float4*)(src + i * 4);
}

// ---------------------------------------------------------------- edge scatter-add
__global__ __launch_bounds__(256) void k_edges(const int* __restrict__ src,
                                               const int* __restrict__ dst,
                                               const float* __restrict__ X,
                                               float* __restrict__ A) {
    int t = blockIdx.x * 256 + threadIdx.x;      // N_EDGES*32 threads
    int e = t >> 5, lane = t & 31;
    if (e < N_EDGES) {
        int s = src[e], d = dst[e];
        float4 v = *(const float4*)(X + (size_t)s * SD + lane * 4);
        float* ap = A + (size_t)d * SD + lane * 4;
        atomicAdd(ap + 0, v.x);
        atomicAdd(ap + 1, v.y);
        atomicAdd(ap + 2, v.z);
        atomicAdd(ap + 3, v.w);
    }
}

// ------------------------------------------------- fused GIN MLP: X = relu(relu(X@W1+b1)@W2+b2), in place
// 8 waves/block; wave w owns rows [blk*128 + 16w, +16). H strip kept in LDS per wave.
// Out-of-range rows: loads clamped to a valid row (their garbage stays confined to
// their own D rows, which are masked at the final store), so the A stage is pure
// b128 loads + packed bf16 converts with no divergence.
__global__ __launch_bounds__(256) void k_gemm_mlp(float* __restrict__ X,
                                                  const __bf16* __restrict__ W1t,
                                                  const float* __restrict__ b1,
                                                  const __bf16* __restrict__ W2t,
                                                  const float* __restrict__ b2) {
    __shared__ __bf16 sH[128 * SD];              // 32 KB, per-wave 16-row strips

    const int tid  = threadIdx.x;
    const int wave = tid >> 5;
    const int lane = tid & 31;
    const int half = lane >> 4;
    const int l16  = lane & 15;
    const int rowBase = blockIdx.x * 128;

    // ---- load A fragments for this wave's rows (K = 0..127, 4 k-steps)
    const int gr  = rowBase + wave * 16 + l16;   // A-matrix row M = l16 (both halves)
    const int grc = gr < N_NODES ? gr : (N_NODES - 1);
    const float* xr = X + (size_t)grc * SD;
    v16bf afrag[4];
#pragma unroll
    for (int kk = 0; kk < 4; ++kk) {
        const int k0 = kk * 32 + half * 16;
        const float4 a0 = *(const float4*)(xr + k0);
        const float4 a1 = *(const float4*)(xr + k0 + 4);
        const float4 a2 = *(const float4*)(xr + k0 + 8);
        const float4 a3 = *(const float4*)(xr + k0 + 12);
        afrag[kk][0]  = (__bf16)a0.x; afrag[kk][1]  = (__bf16)a0.y;
        afrag[kk][2]  = (__bf16)a0.z; afrag[kk][3]  = (__bf16)a0.w;
        afrag[kk][4]  = (__bf16)a1.x; afrag[kk][5]  = (__bf16)a1.y;
        afrag[kk][6]  = (__bf16)a1.z; afrag[kk][7]  = (__bf16)a1.w;
        afrag[kk][8]  = (__bf16)a2.x; afrag[kk][9]  = (__bf16)a2.y;
        afrag[kk][10] = (__bf16)a2.z; afrag[kk][11] = (__bf16)a2.w;
        afrag[kk][12] = (__bf16)a3.x; afrag[kk][13] = (__bf16)a3.y;
        afrag[kk][14] = (__bf16)a3.z; afrag[kk][15] = (__bf16)a3.w;
    }

    // ---- stage 1: H = relu(A @ W1 + b1) -> LDS (bf16)
    for (int n = 0; n < 8; ++n) {
        v8f acc = {};
#pragma unroll
        for (int kk = 0; kk < 4; ++kk) {
            const v16bf b = *(const v16bf*)(W1t + (n * 16 + l16) * SD + kk * 32 + half * 16);
            acc = __builtin_amdgcn_wmma_f32_16x16x32_bf16(
                false, afrag[kk], false, b, (short)0, acc, false, false);
        }
        const int col = n * 16 + l16;
        const float bias = b1[col];
#pragma unroll
        for (int v = 0; v < 8; ++v) {
            float h = acc[v] + bias;
            sH[(wave * 16 + v + 8 * half) * SD + col] = (__bf16)(h > 0.f ? h : 0.f);
        }
    }
    __syncthreads();

    // ---- reload A fragments from H strip (same rows this wave just wrote)
    const __bf16* hr = &sH[(wave * 16 + l16) * SD];
    v16bf hfrag[4];
#pragma unroll
    for (int kk = 0; kk < 4; ++kk) {
        const int k0 = kk * 32 + half * 16;
#pragma unroll
        for (int i = 0; i < 8; ++i) {
            hfrag[kk][i]     = hr[k0 + i];
            hfrag[kk][i + 8] = hr[k0 + 8 + i];
        }
    }

    // ---- stage 2: Y = relu(H @ W2 + b2) -> global (in place, rows owned by this wave)
    for (int n = 0; n < 8; ++n) {
        v8f acc = {};
#pragma unroll
        for (int kk = 0; kk < 4; ++kk) {
            const v16bf b = *(const v16bf*)(W2t + (n * 16 + l16) * SD + kk * 32 + half * 16);
            acc = __builtin_amdgcn_wmma_f32_16x16x32_bf16(
                false, hfrag[kk], false, b, (short)0, acc, false, false);
        }
        const int col = n * 16 + l16;
        const float bias = b2[col];
#pragma unroll
        for (int v = 0; v < 8; ++v) {
            const int r = rowBase + wave * 16 + v + 8 * half;
            if (r < N_NODES) {
                float y = acc[v] + bias;
                X[(size_t)r * SD + col] = y > 0.f ? y : 0.f;
            }
        }
    }
}

// ---------------------------------------------------------------- BN stats (biased var)
__global__ __launch_bounds__(256) void k_bnstats(const float* __restrict__ X,
                                                 float* __restrict__ mean,
                                                 float* __restrict__ rstd) {
    const int c = blockIdx.x;                    // 128 blocks, one per column
    const int tid = threadIdx.x;
    float s = 0.f, q = 0.f;
    for (int r = tid; r < N_NODES; r += 256) {
        float v = X[(size_t)r * SD + c];
        s += v; q += v * v;
    }
    __shared__ float ss[256], sq[256];
    ss[tid] = s; sq[tid] = q;
    __syncthreads();
    for (int o = 128; o > 0; o >>= 1) {
        if (tid < o) { ss[tid] += ss[tid + o]; sq[tid] += sq[tid + o]; }
        __syncthreads();
    }
    if (tid == 0) {
        float m = ss[0] / (float)N_NODES;
        float v = sq[0] / (float)N_NODES - m * m;
        mean[c] = m;
        rstd[c] = rsqrtf(v + 1e-5f);
    }
}

__global__ __launch_bounds__(256) void k_bnapply(float* __restrict__ X,
                                                 const float* __restrict__ mean,
                                                 const float* __restrict__ rstd,
                                                 const float* __restrict__ g,
                                                 const float* __restrict__ b) {
    int i = blockIdx.x * 256 + threadIdx.x;      // N_NODES*SD threads
    if (i < N_NODES * SD) {
        int c = i & 127;
        X[i] = g[c] * (X[i] - mean[c]) * rstd[c] + b[c];
    }
}

// ---------------------------------------------------------------- pooling
__global__ __launch_bounds__(256) void k_zero(float* __restrict__ p, int n) {
    int i = blockIdx.x * 256 + threadIdx.x;
    if (i < n) p[i] = 0.f;
}

__global__ __launch_bounds__(256) void k_pool(const int* __restrict__ batch,
                                              const float* __restrict__ X,
                                              float* __restrict__ pooled) {
    int t = blockIdx.x * 256 + threadIdx.x;      // N_NODES*32 threads
    int node = t >> 5, lane = t & 31;
    if (node < N_NODES) {
        int g = batch[node];
        float4 v = *(const float4*)(X + (size_t)node * SD + lane * 4);
        float* pp = pooled + (size_t)g * SD + lane * 4;
        atomicAdd(pp + 0, v.x);
        atomicAdd(pp + 1, v.y);
        atomicAdd(pp + 2, v.z);
        atomicAdd(pp + 3, v.w);
    }
}

// ---------------------------------------------------------------- FC head (tiny)
__global__ __launch_bounds__(128) void k_fc(const float* __restrict__ pooled,
                                            const float* __restrict__ W1,
                                            const float* __restrict__ b1,
                                            const float* __restrict__ W2,
                                            const float* __restrict__ b2,
                                            float* __restrict__ out) {
    const int g = blockIdx.x;                    // 256 blocks
    const int j = threadIdx.x;                   // 128 threads
    __shared__ float p[SD], h[SD];
    p[j] = pooled[g * SD + j];
    __syncthreads();
    float s = b1[j];
    for (int k = 0; k < SD; ++k) s += p[k] * W1[k * SD + j];
    h[j] = s > 0.f ? s : 0.f;
    __syncthreads();
    if (j < N_CLASS) {
        float o = b2[j];
        for (int k = 0; k < SD; ++k) o += h[k] * W2[k * N_CLASS + j];
        out[g * N_CLASS + j] = o;
    }
}

// ================================================================ launcher
extern "C" void kernel_launch(void* const* d_in, const int* in_sizes, int n_in,
                              void* d_out, int out_size, void* d_ws, size_t ws_size,
                              hipStream_t stream) {
    const int*   tok   = (const int*)d_in[0];
    const int*   esrc  = (const int*)d_in[1];
    const int*   edst  = esrc + N_EDGES;
    const int*   batch = (const int*)d_in[2];
    const float* emb   = (const float*)d_in[3];

    // per-layer (W1, b1, W2, b2, gamma, beta); layer 3 has no BN
    const float* W1[4] = { (const float*)d_in[4],  (const float*)d_in[10],
                           (const float*)d_in[10] + SD * SD, (const float*)d_in[16] };
    const float* B1[4] = { (const float*)d_in[5],  (const float*)d_in[11],
                           (const float*)d_in[11] + SD, (const float*)d_in[17] };
    const float* W2[4] = { (const float*)d_in[6],  (const float*)d_in[12],
                           (const float*)d_in[12] + SD * SD, (const float*)d_in[18] };
    const float* B2[4] = { (const float*)d_in[7],  (const float*)d_in[13],
                           (const float*)d_in[13] + SD, (const float*)d_in[19] };
    const float* GM[3] = { (const float*)d_in[8],  (const float*)d_in[14],
                           (const float*)d_in[14] + SD };
    const float* BT[3] = { (const float*)d_in[9],  (const float*)d_in[15],
                           (const float*)d_in[15] + SD };
    const float* Wfc1 = (const float*)d_in[20];
    const float* bfc1 = (const float*)d_in[21];
    const float* Wfc2 = (const float*)d_in[22];
    const float* bfc2 = (const float*)d_in[23];

    // ---- workspace carve-up
    char* ws = (char*)d_ws;
    size_t off = 0;
    float* xbuf = (float*)(ws + off); off += (size_t)N_NODES * SD * 4;   // 20.48 MB
    float* abuf = (float*)(ws + off); off += (size_t)N_NODES * SD * 4;   // 20.48 MB
    __bf16* wt[8];
    for (int i = 0; i < 8; ++i) { wt[i] = (__bf16*)(ws + off); off += (size_t)SD * SD * 2; }
    float* meanb  = (float*)(ws + off); off += SD * 4;
    float* rstdb  = (float*)(ws + off); off += SD * 4;
    float* pooled = (float*)(ws + off); off += (size_t)N_GRAPHS * SD * 4;

    // ---- weights -> bf16 transposed
    const float* wsrc[8] = { W1[0], W2[0], W1[1], W2[1], W1[2], W2[2], W1[3], W2[3] };
    for (int i = 0; i < 8; ++i)
        k_wconv<<<(SD * SD + 255) / 256, 256, 0, stream>>>(wsrc[i], wt[i]);

    // ---- embedding gather
    k_embed<<<(N_NODES * 32 + 255) / 256, 256, 0, stream>>>(tok, emb, xbuf);

    const int gemmBlocks = (N_NODES + 127) / 128;
    // ---- 4 GIN layers
    for (int l = 0; l < 4; ++l) {
        k_copy<<<(N_NODES * SD / 4 + 255) / 256, 256, 0, stream>>>(xbuf, abuf);
        k_edges<<<(N_EDGES * 32 + 255) / 256, 256, 0, stream>>>(esrc, edst, xbuf, abuf);
        k_gemm_mlp<<<gemmBlocks, 256, 0, stream>>>(abuf, wt[2 * l], B1[l], wt[2 * l + 1], B2[l]);
        if (l < 3) {
            k_bnstats<<<SD, 256, 0, stream>>>(abuf, meanb, rstdb);
            k_bnapply<<<(N_NODES * SD + 255) / 256, 256, 0, stream>>>(abuf, meanb, rstdb, GM[l], BT[l]);
        }
        float* t = xbuf; xbuf = abuf; abuf = t;  // ping-pong
    }

    // ---- graph pooling + FC head
    k_zero<<<(N_GRAPHS * SD + 255) / 256, 256, 0, stream>>>(pooled, N_GRAPHS * SD);
    k_pool<<<(N_NODES * 32 + 255) / 256, 256, 0, stream>>>(batch, xbuf, pooled);
    k_fc<<<N_GRAPHS, 128, 0, stream>>>(pooled, Wfc1, bfc1, Wfc2, bfc2, (float*)d_out);
}
